// GAT_77627238908255
// MI455X (gfx1250) — compile-verified, matrix operand order checked by reading, not run on hardware
//
#include <hip/hip_runtime.h>

typedef __attribute__((ext_vector_type(16))) __bf16 v16bf;
typedef __attribute__((ext_vector_type(8)))  __bf16 v8bf;
typedef __attribute__((ext_vector_type(8)))  float  v8f;

#define CH 256      // H*D
#define HD 64       // D per head
#define NH 4        // heads

// ---------------------------------------------------------------------------
// One-time per layer: Wt[n][k] = (bf16) W[k][n]  (256x256, 128 KB, L2-resident)
// ---------------------------------------------------------------------------
__global__ void transpose_w_bf16(const float* __restrict__ W,
                                 __bf16* __restrict__ Wt) {
  int i = blockIdx.x * blockDim.x + threadIdx.x;  // 65536
  int n = i >> 8, k = i & 255;
  Wt[(size_t)n * CH + k] = (__bf16)W[(size_t)k * CH + n];
}

// ---------------------------------------------------------------------------
// GEMM: C[M,256] = A[M,256] @ W, W pre-transposed to bf16 (Wt[n][k]).
// One wave computes a 16(M) x 64(N) strip: 4 accumulators share one A frag.
// K fully unrolled: 8 steps x 4 WMMA.
// ---------------------------------------------------------------------------
__global__ __launch_bounds__(256) void gemm_wmma_bf16(
    const float* __restrict__ A, const __bf16* __restrict__ Wt,
    float* __restrict__ C, int M, int wavesTotal) {
  int wave = threadIdx.x >> 5;
  int lane = threadIdx.x & 31;
  int w = blockIdx.x * 8 + wave;
  if (w >= wavesTotal) return;
  int tm = w >> 2;             // 16-row tile index
  int strip = w & 3;           // 64-col strip
  int row = tm * 16 + (lane & 15);
  if (row >= M) row = M - 1;   // safe clamp (M % 16 == 0 here)
  int khalf = (lane >> 4) * 8; // lanes 0-15 -> K 0..7/16..23; lanes 16-31 -> +8

  v8f acc[4] = {{}, {}, {}, {}};
  const float* Arow = A + (size_t)row * CH;

#pragma unroll
  for (int k0 = 0; k0 < CH; k0 += 32) {
    // ---- A fragment (16x32 bf16): 4x float4 loads + pk converts
    float4 a0 = *(const float4*)(Arow + k0 + khalf);
    float4 a1 = *(const float4*)(Arow + k0 + khalf + 4);
    float4 a2 = *(const float4*)(Arow + k0 + khalf + 16);
    float4 a3 = *(const float4*)(Arow + k0 + khalf + 20);
    v16bf a;
    a[0] = (__bf16)a0.x;  a[1] = (__bf16)a0.y;
    a[2] = (__bf16)a0.z;  a[3] = (__bf16)a0.w;
    a[4] = (__bf16)a1.x;  a[5] = (__bf16)a1.y;
    a[6] = (__bf16)a1.z;  a[7] = (__bf16)a1.w;
    a[8] = (__bf16)a2.x;  a[9] = (__bf16)a2.y;
    a[10] = (__bf16)a2.z; a[11] = (__bf16)a2.w;
    a[12] = (__bf16)a3.x; a[13] = (__bf16)a3.y;
    a[14] = (__bf16)a3.z; a[15] = (__bf16)a3.w;

    // ---- 4 B fragments (32x16 bf16 each): two b128 loads each, no cvt
#pragma unroll
    for (int s = 0; s < 4; ++s) {
      int col = strip * 64 + s * 16 + (lane & 15);
      const __bf16* wrow = Wt + (size_t)col * CH + k0 + khalf;
      v8bf blo = *(const v8bf*)wrow;
      v8bf bhi = *(const v8bf*)(wrow + 16);
      v16bf b = __builtin_shufflevector(blo, bhi, 0, 1, 2, 3, 4, 5, 6, 7,
                                        8, 9, 10, 11, 12, 13, 14, 15);
      acc[s] = __builtin_amdgcn_wmma_f32_16x16x32_bf16(
          false, a, false, b, (short)0, acc[s], false, false);
    }
  }

  // ---- store: VGPR j -> row (lane>>4)*8 + j, col = lane&15
  int crow = tm * 16 + (lane >> 4) * 8;
  if (crow + 8 <= M) {   // full tile: unconditional stores
#pragma unroll
    for (int s = 0; s < 4; ++s) {
      float* cp = C + (size_t)crow * CH + strip * 64 + s * 16 + (lane & 15);
#pragma unroll
      for (int j = 0; j < 8; ++j) cp[(size_t)j * CH] = acc[s][j];
    }
  } else {
#pragma unroll
    for (int s = 0; s < 4; ++s) {
      float* cp = C + (size_t)crow * CH + strip * 64 + s * 16 + (lane & 15);
#pragma unroll
      for (int j = 0; j < 8; ++j)
        if (crow + j < M) cp[(size_t)j * CH] = acc[s][j];
    }
  }
}

// ---------------------------------------------------------------------------
// el/er: one wave per (n,h); lane-parallel dot + wave32 shuffle reduction.
// ---------------------------------------------------------------------------
__global__ __launch_bounds__(256) void compute_lr(
    const float* __restrict__ F, const float* __restrict__ al,
    const float* __restrict__ ar, float* __restrict__ el,
    float* __restrict__ er, int NHtot) {
  int wv = blockIdx.x * 8 + (threadIdx.x >> 5);
  int lane = threadIdx.x & 31;
  if (wv >= NHtot) return;
  int h = wv & (NH - 1);
  const float* f = F + (size_t)wv * HD;  // n*256 + h*64 == wv*64
  float v0 = f[lane], v1 = f[lane + 32];
  float sl = v0 * al[h * HD + lane] + v1 * al[h * HD + lane + 32];
  float sr = v0 * ar[h * HD + lane] + v1 * ar[h * HD + lane + 32];
#pragma unroll
  for (int off = 16; off > 0; off >>= 1) {
    sl += __shfl_xor(sl, off);
    sr += __shfl_xor(sr, off);
  }
  if (lane == 0) {
    el[wv] = sl;
    er[wv] = sr;
  }
}

__global__ void fill_kernel(float* __restrict__ p, float v, int n) {
  int i = blockIdx.x * blockDim.x + threadIdx.x;
  if (i < n) p[i] = v;
}

// rst = input (residual) + bias
__global__ void init_rst(const float* __restrict__ in,
                         const float* __restrict__ bias,
                         float* __restrict__ R, int total) {
  int i = blockIdx.x * blockDim.x + threadIdx.x;
  if (i >= total) return;
  R[i] = in[i] + bias[i & (CH - 1)];
}

__device__ inline void atomicMaxFloat(float* addr, float val) {
  if (val >= 0.f) atomicMax((int*)addr, __float_as_int(val));
  else            atomicMin((unsigned int*)addr, __float_as_uint(val));
}

// e[e,h] = leaky_relu(el[src]+er[dst], 0.2); atomic segment-max into m[dst,h]
__global__ void edge_logits(const int* __restrict__ src,
                            const int* __restrict__ dst,
                            const float* __restrict__ el,
                            const float* __restrict__ er,
                            float* __restrict__ ew, float* __restrict__ m,
                            int E) {
  int i = blockIdx.x * blockDim.x + threadIdx.x;
  if (i >= E) return;
  int s = src[i], d = dst[i];
#pragma unroll
  for (int h = 0; h < NH; ++h) {
    float v = el[s * NH + h] + er[d * NH + h];
    v = v > 0.f ? v : 0.2f * v;
    ew[i * NH + h] = v;
    atomicMaxFloat(&m[d * NH + h], v);
  }
}

// ee = exp(e - m[dst]) in place; denom[dst,h] += ee
__global__ void edge_exp(const int* __restrict__ dst,
                         const float* __restrict__ m,
                         float* __restrict__ ew, float* __restrict__ dn,
                         int E) {
  int i = blockIdx.x * blockDim.x + threadIdx.x;
  if (i >= E) return;
  int d = dst[i];
#pragma unroll
  for (int h = 0; h < NH; ++h) {
    float v = __expf(ew[i * NH + h] - m[d * NH + h]);
    ew[i * NH + h] = v;
    atomicAdd(&dn[d * NH + h], v);
  }
}

// One block (256 threads) per edge: rst[dst,:] += feat[src,:] * alpha[h]
__global__ __launch_bounds__(256) void edge_aggregate(
    const int* __restrict__ src, const int* __restrict__ dst,
    const float* __restrict__ F, const float* __restrict__ ew,
    const float* __restrict__ dn, float* __restrict__ R) {
  int e = blockIdx.x;
  int t = threadIdx.x;
  int s = src[e], d = dst[e];
  __shared__ float alpha[NH];
  if (t < NH) {
    float den = dn[d * NH + t];
    alpha[t] = ew[e * NH + t] / fmaxf(den, 1e-9f);
  }
  __syncthreads();
  float msg = F[(size_t)s * CH + t] * alpha[t >> 6];
  atomicAdd(&R[(size_t)d * CH + t], msg);
}

// leaky_relu(0.01) in place + per-channel BN partial sums (thread = channel)
__global__ __launch_bounds__(256) void act_stats(float* __restrict__ R,
                                                 float* __restrict__ sums,
                                                 int Nn) {
  int t = threadIdx.x;             // channel
  int n0 = blockIdx.x * 256;       // node slab
  float s = 0.f, s2 = 0.f;
  for (int r = 0; r < 256; ++r) {
    int n = n0 + r;
    if (n >= Nn) break;
    float v = R[(size_t)n * CH + t];
    v = v > 0.f ? v : 0.01f * v;
    R[(size_t)n * CH + t] = v;
    s += v;
    s2 += v * v;
  }
  atomicAdd(&sums[t], s);
  atomicAdd(&sums[CH + t], s2);
}

__global__ void bn_params(const float* __restrict__ sums,
                          const float* __restrict__ g,
                          const float* __restrict__ be,
                          float* __restrict__ sc, float* __restrict__ sh,
                          float invN) {
  int c = threadIdx.x;
  float mu = sums[c] * invN;
  float var = sums[CH + c] * invN - mu * mu;
  float s = g[c] * rsqrtf(var + 1e-5f);
  sc[c] = s;
  sh[c] = be[c] - mu * s;
}

__global__ void bn_apply(float* __restrict__ R, const float* __restrict__ sc,
                         const float* __restrict__ sh, int total) {
  int i = blockIdx.x * blockDim.x + threadIdx.x;
  if (i >= total) return;
  int c = i & (CH - 1);
  R[i] = R[i] * sc[c] + sh[c];
}

__global__ void head_mean(const float* __restrict__ R, float* __restrict__ out,
                          int Nn) {
  int i = blockIdx.x * blockDim.x + threadIdx.x;
  if (i >= Nn * HD) return;
  int n = i >> 6, d = i & (HD - 1);
  const float* r = R + (size_t)n * CH;
  out[i] = 0.25f * (r[d] + r[HD + d] + r[2 * HD + d] + r[3 * HD + d]);
}

// ---------------------------------------------------------------------------
static void run_gat_layer(const float* in, const float* W, const float* al,
                          const float* ar, const float* bias, const int* src,
                          const int* dst, int N, int E, float* F, float* R,
                          float* el, float* er, float* m, float* dn, float* ew,
                          __bf16* Wt, hipStream_t stream) {
  transpose_w_bf16<<<(CH * CH) / 256, 256, 0, stream>>>(W, Wt);

  int rowTiles = (N + 15) / 16;
  int wavesTotal = rowTiles * 4;              // 4 x 64-col strips
  int gemmBlocks = (wavesTotal + 7) / 8;
  gemm_wmma_bf16<<<gemmBlocks, 256, 0, stream>>>(in, Wt, F, N, wavesTotal);

  int NHtot = N * NH;
  compute_lr<<<(NHtot + 7) / 8, 256, 0, stream>>>(F, al, ar, el, er, NHtot);

  fill_kernel<<<(NHtot + 255) / 256, 256, 0, stream>>>(m, -1e30f, NHtot);
  fill_kernel<<<(NHtot + 255) / 256, 256, 0, stream>>>(dn, 0.f, NHtot);

  int total = N * CH;
  init_rst<<<(total + 255) / 256, 256, 0, stream>>>(in, bias, R, total);

  edge_logits<<<(E + 255) / 256, 256, 0, stream>>>(src, dst, el, er, ew, m, E);
  edge_exp<<<(E + 255) / 256, 256, 0, stream>>>(dst, m, ew, dn, E);
  edge_aggregate<<<E, 256, 0, stream>>>(src, dst, F, ew, dn, R);
}

static void run_bn(float* R, const float* g, const float* be, float* bn,
                   int N, hipStream_t stream) {
  float* sums = bn;            // [0..511] sum / sumsq
  float* sc = bn + 2 * CH;     // [512..767]
  float* sh = bn + 3 * CH;     // [768..1023]
  fill_kernel<<<(2 * CH + 255) / 256, 256, 0, stream>>>(sums, 0.f, 2 * CH);
  act_stats<<<(N + 255) / 256, 256, 0, stream>>>(R, sums, N);
  bn_params<<<1, CH, 0, stream>>>(sums, g, be, sc, sh, 1.0f / (float)N);
  bn_apply<<<(N * CH + 255) / 256, 256, 0, stream>>>(R, sc, sh, N * CH);
}

extern "C" void kernel_launch(void* const* d_in, const int* in_sizes, int n_in,
                              void* d_out, int out_size, void* d_ws,
                              size_t ws_size, hipStream_t stream) {
  const float* x   = (const float*)d_in[0];
  const int*   src = (const int*)d_in[1];
  const int*   dst = (const int*)d_in[2];
  const float* W1  = (const float*)d_in[3];
  const float* al1 = (const float*)d_in[4];
  const float* ar1 = (const float*)d_in[5];
  const float* b1  = (const float*)d_in[6];
  const float* W2  = (const float*)d_in[7];
  const float* al2 = (const float*)d_in[8];
  const float* ar2 = (const float*)d_in[9];
  const float* b2  = (const float*)d_in[10];
  const float* W3  = (const float*)d_in[11];
  const float* al3 = (const float*)d_in[12];
  const float* ar3 = (const float*)d_in[13];
  const float* b3  = (const float*)d_in[14];
  const float* g1  = (const float*)d_in[15];
  const float* be1 = (const float*)d_in[16];
  const float* g2  = (const float*)d_in[17];
  const float* be2 = (const float*)d_in[18];

  int N = in_sizes[0] / CH;
  int E = in_sizes[1];
  float* out = (float*)d_out;

  // workspace carve
  size_t NC = (size_t)N * CH;
  float* ws = (float*)d_ws;
  float* B0 = ws;                 // feat buffer (all layers)
  float* B1 = B0 + NC;            // rst L1 -> input L2; rst L3
  float* B2 = B1 + NC;            // rst L2 -> input L3
  float* el = B2 + NC;
  float* er = el + (size_t)N * NH;
  float* m  = er + (size_t)N * NH;
  float* dn = m  + (size_t)N * NH;
  float* ew = dn + (size_t)N * NH;        // [E,4] logits / exp (in-place)
  float* bn = ew + (size_t)E * NH;        // 1024 floats BN scratch
  __bf16* Wt = (__bf16*)(bn + 1024);      // 256*256 bf16 transposed weight

  // Layer 1: x -> B1 (act + BN)
  run_gat_layer(x, W1, al1, ar1, b1, src, dst, N, E, B0, B1, el, er, m, dn, ew,
                Wt, stream);
  run_bn(B1, g1, be1, bn, N, stream);

  // Layer 2: B1 -> B2 (act + BN)
  run_gat_layer(B1, W2, al2, ar2, b2, src, dst, N, E, B0, B2, el, er, m, dn, ew,
                Wt, stream);
  run_bn(B2, g2, be2, bn, N, stream);

  // Layer 3: B2 -> B1 (no act/BN), then head mean -> out [N,64]
  run_gat_layer(B2, W3, al3, ar3, b3, src, dst, N, E, B0, B1, el, er, m, dn, ew,
                Wt, stream);
  head_mean<<<(N * HD + 255) / 256, 256, 0, stream>>>(B1, out, N);
}
